// GATv2_85323820303042
// MI455X (gfx1250) — matrix operand order repaired and checked by reading, not compiled, not run on hardware
//
#include <hip/hip_runtime.h>
#include <hip/hip_bf16.h>
#include <stdint.h>

#define N_NODES 50000
#define N_EDGES 800000
#define D_FEAT  128
#define N_HEADS 4
#define HEAD_DIM 32
#define E_TILES (N_EDGES / 16)
#define N_TILES_NODE (N_NODES / 16)

typedef __bf16 bf16_t;
typedef bf16_t v16bf __attribute__((ext_vector_type(16)));
typedef float  v16f  __attribute__((ext_vector_type(16)));
typedef float  v8f   __attribute__((ext_vector_type(8)));

union BFVecLds { uint4 q[2]; v16bf v; };
union F16Vec   { float4 f4[4]; v16f v; };

// ---- helpers ---------------------------------------------------------------
__device__ inline uint16_t f2u16bf(float f) {
  return __builtin_bit_cast(uint16_t, (bf16_t)f);      // v_cvt to bf16 (RNE)
}
__device__ inline float bfu2f(uint32_t u16) { return __uint_as_float(u16 << 16); }

// order-preserving f32 <-> u32 encoding for atomicMax
__device__ inline unsigned f2ord(float f) {
  unsigned u = __float_as_uint(f);
  return (u & 0x80000000u) ? ~u : (u | 0x80000000u);
}
__device__ inline float ord2f(unsigned u) {
  unsigned v = (u & 0x80000000u) ? (u & 0x7FFFFFFFu) : ~u;
  return __uint_as_float(v);
}

// mish(t) = t * tanh(softplus(t)) = t * g/(g+2), g = e*(e+2), e = exp(t)
__device__ inline float mish(float t) {
  float e = __expf(t);
  float g = e * (e + 2.0f);
  float m = t * __fdividef(g, g + 2.0f);
  return (t > 20.0f) ? t : m;
}

// A fragment (16x32 bf16): this lane's row, K = kb..kb+7 in v0-3, kb+16..+23 in v4-7
__device__ inline v16bf load_a_frag(const float* __restrict__ row, int kb) {
  F16Vec t;
  t.f4[0] = *reinterpret_cast<const float4*>(row + kb);
  t.f4[1] = *reinterpret_cast<const float4*>(row + kb + 4);
  t.f4[2] = *reinterpret_cast<const float4*>(row + kb + 16);
  t.f4[3] = *reinterpret_cast<const float4*>(row + kb + 20);
  return __builtin_convertvector(t.v, v16bf);          // v_cvt_pk_bf16_f32 x8
}

// B fragment (32x16 bf16) from LDS weights stored transposed [n][k]
__device__ inline v16bf load_b_frag(const bf16_t* __restrict__ wp) {
  BFVecLds r;
  const uint4* p = reinterpret_cast<const uint4*>(wp);
  r.q[0] = p[0];
  r.q[1] = p[2];
  return r.v;
}

// ---- Kernel P: per-node GEMMs  S = nodes*Ws + (Ws_b+We_b), R = nodes*Wr + Wr_b
__global__ __launch_bounds__(256) void node_gemm_kernel(
    const float* __restrict__ nodes,
    const float* __restrict__ Ws, const float* __restrict__ Ws_b,
    const float* __restrict__ Wr, const float* __restrict__ Wr_b,
    const float* __restrict__ We_b,
    float* __restrict__ S_out, float* __restrict__ R_out)
{
  __shared__ bf16_t wtS[128 * 128];     // transposed [n][k] bf16, 32KB each
  __shared__ bf16_t wtR[128 * 128];
  __shared__ float  be_s[128], br_s[128];

  const int tid = threadIdx.x;
  for (int i = tid; i < 128 * 128; i += 256) {
    int k = i >> 7, n = i & 127;
    wtS[n * 128 + k] = (bf16_t)Ws[i];
    wtR[n * 128 + k] = (bf16_t)Wr[i];
  }
  if (tid < 128) { be_s[tid] = Ws_b[tid] + We_b[tid]; br_s[tid] = Wr_b[tid]; }
  __syncthreads();

  const int lane = tid & 31, wid = tid >> 5;
  const int half = lane >> 4, nloc = lane & 15;
  const int nwaves = gridDim.x * 8;
  const bf16_t* wpS = wtS + nloc * 128 + half * 8;
  const bf16_t* wpR = wtR + nloc * 128 + half * 8;

  for (int tile = blockIdx.x * 8 + wid; tile < N_TILES_NODE; tile += nwaves) {
    const int n0 = tile * 16;
    const float* xrow = nodes + (size_t)(n0 + nloc) * 128 + half * 8;

    v16bf a[4];
    #pragma unroll
    for (int ks = 0; ks < 4; ++ks) a[ks] = load_a_frag(xrow, ks * 32);

    #pragma unroll
    for (int nt = 0; nt < 8; ++nt) {
      v8f accS = {}, accR = {};
      #pragma unroll
      for (int ks = 0; ks < 4; ++ks) {
        v16bf bS = load_b_frag(wpS + nt * 2048 + ks * 32);
        v16bf bR = load_b_frag(wpR + nt * 2048 + ks * 32);
        accS = __builtin_amdgcn_wmma_f32_16x16x32_bf16(false, a[ks], false, bS,
                                                       (short)0, accS, false, false);
        accR = __builtin_amdgcn_wmma_f32_16x16x32_bf16(false, a[ks], false, bR,
                                                       (short)0, accR, false, false);
      }
      const int n = nt * 16 + nloc;
      const float be = be_s[n], br = br_s[n];
      #pragma unroll
      for (int v = 0; v < 8; ++v) {
        const int m = v + half * 8;
        S_out[(size_t)(n0 + m) * 128 + n] = accS[v] + be;
        R_out[(size_t)(n0 + m) * 128 + n] = accR[v] + br;
      }
    }
  }
}

// ---- Kernel A: per-edge We GEMM (WMMA) + gather S/R + mish + logits --------
__global__ __launch_bounds__(256) void edge_kernel(
    const float* __restrict__ edges,
    const float* __restrict__ We,
    const float* __restrict__ attn_w, const float* __restrict__ attn_b,
    const int* __restrict__ senders, const int* __restrict__ receivers,
    const float* __restrict__ S, const float* __restrict__ R,
    uint16_t* __restrict__ eatt_out, float* __restrict__ logits_out)
{
  __shared__ bf16_t wtE[128 * 128];     // transposed [n][k] bf16, 32KB
  __shared__ float  aw_s[32];

  const int tid = threadIdx.x;
  for (int i = tid; i < 128 * 128; i += 256) {
    int k = i >> 7, n = i & 127;
    wtE[n * 128 + k] = (bf16_t)We[i];
  }
  if (tid < 32) aw_s[tid] = attn_w[tid];
  __syncthreads();

  const int lane = tid & 31, wid = tid >> 5;
  const int half = lane >> 4, nloc = lane & 15;
  const int nwaves = gridDim.x * 8;
  const float ab  = attn_b[0];
  const bf16_t* wpE = wtE + nloc * 128 + half * 8;

  for (int tile = blockIdx.x * 8 + wid; tile < E_TILES; tile += nwaves) {
    const int e0 = tile * 16;
    const int e  = e0 + nloc;
    const int sidx = senders[e];
    const int ridx = receivers[e];
    const float* xrow = edges + (size_t)e * 128 + half * 8;

    // broadcast per-row S/R offsets to all lanes (row m = v + half*8)
    int soff[8], roff[8];
    #pragma unroll
    for (int v = 0; v < 8; ++v) {
      soff[v] = __shfl(sidx, v + half * 8, 32) * 128;
      roff[v] = __shfl(ridx, v + half * 8, 32) * 128;
    }

    v16bf aE[4];
    #pragma unroll
    for (int ks = 0; ks < 4; ++ks) aE[ks] = load_a_frag(xrow, ks * 32);

    const float aw0 = aw_s[nloc], aw1 = aw_s[16 + nloc];
    float c0 = 0.f, c1 = 0.f;     // captured logits (2 per lane)

    #pragma unroll
    for (int h = 0; h < 4; ++h) {
      float ph[8] = {0.f, 0.f, 0.f, 0.f, 0.f, 0.f, 0.f, 0.f};
      #pragma unroll
      for (int sub = 0; sub < 2; ++sub) {
        const int nt = h * 2 + sub;
        v8f accE = {};
        #pragma unroll
        for (int ks = 0; ks < 4; ++ks) {
          v16bf bE = load_b_frag(wpE + nt * 2048 + ks * 32);
          accE = __builtin_amdgcn_wmma_f32_16x16x32_bf16(false, aE[ks], false, bE,
                                                         (short)0, accE, false, false);
        }
        const int n = nt * 16 + nloc;
        const float aw = sub ? aw1 : aw0;
        float Sv[8], Rv[8];
        #pragma unroll
        for (int v = 0; v < 8; ++v) { Sv[v] = S[soff[v] + n]; Rv[v] = R[roff[v] + n]; }
        #pragma unroll
        for (int v = 0; v < 8; ++v) {
          const int m = v + half * 8;
          float s = accE[v] + Sv[v];            // e_att element (biases folded in S)
          float t = s + Rv[v];
          ph[v] += mish(t) * aw;
          eatt_out[(size_t)(e0 + m) * 128 + n] = f2u16bf(s);
        }
      }
      #pragma unroll
      for (int v = 0; v < 8; ++v) {
        float val = ph[v];
        #pragma unroll
        for (int off = 8; off; off >>= 1) val += __shfl_xor(val, off, 16);
        const int idx = h * 8 + v;              // all 16 lanes now hold the sum
        c0 = (idx == nloc)      ? val : c0;
        c1 = (idx == nloc + 16) ? val : c1;
      }
    }
    // c0 -> (m = (nloc&7)+half*8, h = nloc>>3); c1 -> same m, h + 2
    const int m  = (nloc & 7) + half * 8;
    const int h0 = nloc >> 3;
    logits_out[(size_t)(e0 + m) * 4 + h0]     = c0 + ab;
    logits_out[(size_t)(e0 + m) * 4 + h0 + 2] = c1 + ab;
  }
}

// ---- Init: zero segmax / denom / output ------------------------------------
__global__ void init_kernel(unsigned* __restrict__ segmax, float* __restrict__ denom,
                            float* __restrict__ out) {
  int i = blockIdx.x * 256 + threadIdx.x;
  if (i < N_NODES * 4) { segmax[i] = 0u; denom[i] = 0.f; }
  if (i < N_NODES * 128) out[i] = 0.f;
}

// ---- B: segment max via order-preserving uint atomicMax --------------------
__global__ void segmax_kernel(const float* __restrict__ logits,
                              const int* __restrict__ receivers,
                              unsigned* __restrict__ segmax) {
  int i = blockIdx.x * 256 + threadIdx.x;
  if (i >= N_EDGES * 4) return;
  int e = i >> 2, h = i & 3;
  atomicMax(&segmax[receivers[e] * 4 + h], f2ord(logits[i]));
}

// ---- C: w = exp(logit - max), denom += w -----------------------------------
__global__ void expsum_kernel(float* __restrict__ logits,
                              const int* __restrict__ receivers,
                              const unsigned* __restrict__ segmax,
                              float* __restrict__ denom) {
  int i = blockIdx.x * 256 + threadIdx.x;
  if (i >= N_EDGES * 4) return;
  int e = i >> 2, h = i & 3;
  int r = receivers[e];
  float w = __expf(logits[i] - ord2f(segmax[r * 4 + h]));
  logits[i] = w;
  atomicAdd(&denom[r * 4 + h], w);
}

// ---- D: normalized weighted scatter of e_att into out ----------------------
__global__ __launch_bounds__(256) void scatter_kernel(
    const uint16_t* __restrict__ eatt, const float* __restrict__ w,
    const float* __restrict__ denom, const int* __restrict__ receivers,
    float* __restrict__ out) {
  long long gid = (long long)blockIdx.x * 256 + threadIdx.x;
  int e = (int)(gid >> 5);
  if (e >= N_EDGES) return;
  int j = (int)(gid & 31);        // wave32: one wave per edge
  int d = j * 4;
  int h = j >> 3;                 // 4 elems per lane stay within one head
  int r = receivers[e];
  float c = w[(size_t)e * 4 + h] / denom[r * 4 + h];
  uint2 q = *reinterpret_cast<const uint2*>(eatt + (size_t)e * 128 + d);
  float* po = out + (size_t)r * 128 + d;
  atomicAdd(po + 0, c * bfu2f(q.x & 0xFFFFu));
  atomicAdd(po + 1, c * bfu2f(q.x >> 16));
  atomicAdd(po + 2, c * bfu2f(q.y & 0xFFFFu));
  atomicAdd(po + 3, c * bfu2f(q.y >> 16));
}

// ---- launch ----------------------------------------------------------------
extern "C" void kernel_launch(void* const* d_in, const int* in_sizes, int n_in,
                              void* d_out, int out_size, void* d_ws, size_t ws_size,
                              hipStream_t stream) {
  const float* nodes = (const float*)d_in[0];
  const float* edges = (const float*)d_in[1];
  const float* Ws    = (const float*)d_in[2];
  const float* Ws_b  = (const float*)d_in[3];
  const float* Wr    = (const float*)d_in[4];
  const float* Wr_b  = (const float*)d_in[5];
  const float* We    = (const float*)d_in[6];
  const float* We_b  = (const float*)d_in[7];
  const float* aw    = (const float*)d_in[8];
  const float* ab    = (const float*)d_in[9];
  const int*   snd   = (const int*)d_in[10];
  const int*   rcv   = (const int*)d_in[11];
  float* out = (float*)d_out;

  char* ws = (char*)d_ws;
  size_t off = 0;
  uint16_t* eatt = (uint16_t*)(ws + off);
  off += (size_t)N_EDGES * 128 * 2; off = (off + 255) & ~(size_t)255;
  float* logits = (float*)(ws + off);
  off += (size_t)N_EDGES * 4 * 4;   off = (off + 255) & ~(size_t)255;
  unsigned* segmax = (unsigned*)(ws + off);
  off += (size_t)N_NODES * 4 * 4;   off = (off + 255) & ~(size_t)255;
  float* denom = (float*)(ws + off);
  off += (size_t)N_NODES * 4 * 4;   off = (off + 255) & ~(size_t)255;
  float* Sbuf = (float*)(ws + off);
  off += (size_t)N_NODES * 128 * 4; off = (off + 255) & ~(size_t)255;
  float* Rbuf = (float*)(ws + off);

  init_kernel<<<(N_NODES * 128 + 255) / 256, 256, 0, stream>>>(segmax, denom, out);
  node_gemm_kernel<<<(N_TILES_NODE + 7) / 8, 256, 0, stream>>>(
      nodes, Ws, Ws_b, Wr, Wr_b, We_b, Sbuf, Rbuf);
  edge_kernel<<<1024, 256, 0, stream>>>(edges, We, aw, ab, snd, rcv,
                                        Sbuf, Rbuf, eatt, logits);
  segmax_kernel<<<(N_EDGES * 4 + 255) / 256, 256, 0, stream>>>(logits, rcv, segmax);
  expsum_kernel<<<(N_EDGES * 4 + 255) / 256, 256, 0, stream>>>(logits, rcv, segmax, denom);
  scatter_kernel<<<(N_EDGES * 32 + 255) / 256, 256, 0, stream>>>(eatt, logits, denom, rcv, out);
}
